// Linear_int8_STE_82617990906235
// MI455X (gfx1250) — 8-bit matrix operand order repaired and checked by reading, compiled, not run on hardware
//
#include <hip/hip_runtime.h>
#include <stdint.h>

typedef __attribute__((ext_vector_type(8))) int v8i;

#define M_DIM 1024
#define K_DIM 1024
#define N_DIM 1024

__global__ void absmax_kernel(const float* __restrict__ x,
                              const float* __restrict__ w,
                              unsigned* __restrict__ amax) {
  __shared__ float red[256];
  const float* src = blockIdx.y ? w : x;
  float m = 0.f;
  for (int i = blockIdx.x * 256 + threadIdx.x; i < M_DIM * K_DIM; i += gridDim.x * 256)
    m = fmaxf(m, fabsf(src[i]));
  red[threadIdx.x] = m;
  __syncthreads();
  for (int s = 128; s > 0; s >>= 1) {
    if (threadIdx.x < s) red[threadIdx.x] = fmaxf(red[threadIdx.x], red[threadIdx.x + s]);
    __syncthreads();
  }
  if (threadIdx.x == 0) atomicMax(&amax[blockIdx.y], __float_as_uint(red[0]));
}

__global__ void quant_kernel(const float* __restrict__ x, const float* __restrict__ w,
                             const unsigned* __restrict__ amax,
                             uint8_t* __restrict__ qxu, uint8_t* __restrict__ qwu) {
  int i = blockIdx.x * 256 + threadIdx.x;
  const float* src = blockIdx.y ? w : x;
  uint8_t* dst = blockIdx.y ? qwu : qxu;
  float inv = 127.0f / __uint_as_float(amax[blockIdx.y]);
  float q = rintf(src[i] * inv);
  q = fminf(fmaxf(q, -128.f), 127.f);
  dst[i] = (uint8_t)(int)(q + 128.f);
}

__launch_bounds__(256)
__global__ void lutmm_kernel(const uint8_t* __restrict__ qxu,
                             const uint8_t* __restrict__ qwu,
                             const float* __restrict__ lut,
                             const float* __restrict__ bias,
                             const unsigned* __restrict__ amax,
                             float* __restrict__ y) {
  extern __shared__ uint8_t smem[];
  _Float16* rlut = (_Float16*)smem;
  uint8_t* axt = smem + 131072;
  uint8_t* bwt = axt + 4096;

  const int tid = threadIdx.x;
  const int bm = blockIdx.y, bn = blockIdx.x;

  for (int i = tid; i < 65536; i += 256) {
    int a = i >> 8, b = i & 255;
    rlut[i] = (_Float16)(lut[i] - (float)((a - 128) * (b - 128)));
  }

  const float scale = (__uint_as_float(amax[0]) * (1.0f / 127.0f)) *
                      (__uint_as_float(amax[1]) * (1.0f / 127.0f));

  const int lane = tid & 31;
  const int wv   = tid >> 5;
  const int wm   = wv & 3;
  const int wn2  = wv >> 2;
  const int hi   = lane >> 4;
  const int ln   = lane & 15;

  v8i c0 = {};
  v8i c1 = {};
  float rs0[8], rs1[8];
#pragma unroll
  for (int j = 0; j < 8; ++j) { rs0[j] = 0.f; rs1[j] = 0.f; }

  const int mrow  = bm * 64 + wm * 16 + ln;
  const int nrow0 = bn * 64 + wn2 * 32 + ln;

  const int tl_m = tid & 63;
  const int tl_k = (tid >> 6) * 16;

  __syncthreads();

  for (int kc = 0; kc < 16; ++kc) {
    const int kb = kc * 64;
    if (kc) __syncthreads();

    {
      const uint8_t* gsrc = qwu + (size_t)(bn * 64 + tl_m) * K_DIM + kb + tl_k;
      unsigned ldsoff = (unsigned)(uintptr_t)(bwt + tl_m * 64 + tl_k);
      unsigned long long ga = (unsigned long long)(uintptr_t)gsrc;
      asm volatile("global_load_async_to_lds_b128 %0, %1, off"
                   :: "v"(ldsoff), "v"(ga) : "memory");
    }
    {
      union { uint4 v; uint8_t b[16]; } xv;
      xv.v = *(const uint4*)(qxu + (size_t)(bm * 64 + tl_m) * K_DIM + kb + tl_k);
#pragma unroll
      for (int j = 0; j < 16; ++j) axt[(tl_k + j) * 64 + tl_m] = xv.b[j];
    }
    asm volatile("s_wait_asynccnt 0" ::: "memory");
    __syncthreads();

    {
      const unsigned* xr  = (const unsigned*)(qxu + (size_t)mrow * K_DIM + kb);
      const unsigned* wr0 = (const unsigned*)(qwu + (size_t)nrow0 * K_DIM + kb);
      const unsigned* wr1 = wr0 + (16 * K_DIM / 4);
      v8i a, b0, b1;
#pragma unroll
      for (int v = 0; v < 8; ++v) {
        a[v]  = (int)(xr [(v >> 2) * 8 + hi * 4 + (v & 3)] ^ 0x80808080u);
        b0[v] = (int)(wr0[(v >> 2) * 8 + hi * 4 + (v & 3)] ^ 0x80808080u);
        b1[v] = (int)(wr1[(v >> 2) * 8 + hi * 4 + (v & 3)] ^ 0x80808080u);
      }
      c0 = __builtin_amdgcn_wmma_i32_16x16x64_iu8(true, a, true, b0, c0, false, false);
      c1 = __builtin_amdgcn_wmma_i32_16x16x64_iu8(true, a, true, b1, c1, false, false);
    }

    if (kc + 1 < 16) {
      __builtin_prefetch(qxu + (size_t)mrow * K_DIM + kb + 64, 0, 1);
      __builtin_prefetch(qwu + (size_t)nrow0 * K_DIM + kb + 64, 0, 1);
    }

    {
      const int mb = wm * 16 + hi * 8;
      const uint8_t* brow0 = bwt + (wn2 * 32 + ln) * 64;
      const uint8_t* brow1 = brow0 + 16 * 64;
      for (int k16 = 0; k16 < 4; ++k16) {
        union { uint4 v; uint8_t b[16]; } q0, q1;
        q0.v = *(const uint4*)(brow0 + k16 * 16);
        q1.v = *(const uint4*)(brow1 + k16 * 16);
        _Float16 t0[2][8], t1[2][8];

#define GATHER_K(kk, buf)                                                        \
        {                                                                        \
          uint2 a8 = *(const uint2*)(axt + (k16 * 16 + (kk)) * 64 + mb);         \
          const _Float16* r0 = rlut + (unsigned)q0.b[kk];                        \
          const _Float16* r1 = rlut + (unsigned)q1.b[kk];                        \
          _Pragma("unroll")                                                      \
          for (int j = 0; j < 8; ++j) {                                          \
            unsigned av = ((j < 4 ? (a8.x >> (8 * j)) : (a8.y >> (8 * (j - 4)))) \
                           & 255u);                                              \
            t0[buf][j] = r0[av << 8];                                            \
            t1[buf][j] = r1[av << 8];                                            \
          }                                                                      \
        }
#define ACCUM_K(buf)                                                             \
        {                                                                        \
          _Pragma("unroll")                                                      \
          for (int j = 0; j < 8; ++j) {                                          \
            rs0[j] += (float)t0[buf][j];                                         \
            rs1[j] += (float)t1[buf][j];                                         \
          }                                                                      \
        }

        GATHER_K(0, 0)
#pragma unroll
        for (int kk = 1; kk < 16; ++kk) {
          GATHER_K(kk, kk & 1)
          ACCUM_K((kk - 1) & 1)
        }
        ACCUM_K(1)
#undef GATHER_K
#undef ACCUM_K
      }
    }
  }

#pragma unroll
  for (int j = 0; j < 8; ++j) {
    int m  = bm * 64 + wm * 16 + hi * 8 + j;
    int n0 = bn * 64 + wn2 * 32 + ln;
    y[(size_t)m * N_DIM + n0]      = ((float)c0[j] + rs0[j]) * scale + bias[n0];
    y[(size_t)m * N_DIM + n0 + 16] = ((float)c1[j] + rs1[j]) * scale + bias[n0 + 16];
  }
}

extern "C" void kernel_launch(void* const* d_in, const int* in_sizes, int n_in,
                              void* d_out, int out_size, void* d_ws, size_t ws_size,
                              hipStream_t stream) {
  const float* x    = (const float*)d_in[0];
  const float* w    = (const float*)d_in[1];
  const float* bias = (const float*)d_in[2];
  const float* lut  = (const float*)d_in[3];
  float* y = (float*)d_out;

  unsigned* amax = (unsigned*)d_ws;
  uint8_t* qxu = (uint8_t*)d_ws + 256;
  uint8_t* qwu = qxu + (size_t)M_DIM * K_DIM;

  hipMemsetAsync(amax, 0, 2 * sizeof(unsigned), stream);
  absmax_kernel<<<dim3(64, 2), 256, 0, stream>>>(x, w, amax);
  quant_kernel<<<dim3(M_DIM * K_DIM / 256, 2), 256, 0, stream>>>(x, w, amax, qxu, qwu);

  const size_t shmem = 131072 + 8192;
  (void)hipFuncSetAttribute((const void*)lutmm_kernel,
                            hipFuncAttributeMaxDynamicSharedMemorySize, (int)shmem);
  lutmm_kernel<<<dim3(N_DIM / 64, M_DIM / 64), 256, shmem, stream>>>(qxu, qwu, lut, bias, amax, y);
}
